// Attention_20504173871184
// MI455X (gfx1250) — compile-verified, matrix operand order checked by reading, not run on hardware
//
#include <hip/hip_runtime.h>
#include <hip/hip_bf16.h>
#include <math.h>

// ---------- shapes ----------
#define DIMD   768
#define NHEAD  12
#define HD     64
#define BATCH  8
#define SEQ    1024
#define MROWS  (BATCH * SEQ)          // 8192
#define NQKV   (3 * DIMD)             // 2304
#define SCALE  0.125f                 // 1/sqrt(64)

// f16 element counts in workspace
#define XH_ELEMS     (MROWS * DIMD)           // 6291456
#define WQKV_ELEMS   (DIMD * NQKV)            // 1769472  (stored transposed: [2304][768])
#define WPROJ_ELEMS  (DIMD * DIMD)            // 589824   (stored transposed: [768][768])
#define HEADBUF      (MROWS * HD * NHEAD)     // 6291456 (== B*H*N*64)
#define QKV_ELEMS    (3 * HEADBUF)            // Q,K: [bh][seq][hd]; V: [bh][hd][seq]

#define OFF_XH     0
#define OFF_WQKV   (OFF_XH + XH_ELEMS)
#define OFF_WPROJ  (OFF_WQKV + WQKV_ELEMS)
#define OFF_QKV    (OFF_WPROJ + WPROJ_ELEMS)
#define OFF_WA     (OFF_QKV + QKV_ELEMS)

typedef _Float16 h16;
typedef __attribute__((ext_vector_type(4)))  _Float16 v4h;
typedef __attribute__((ext_vector_type(8)))  _Float16 v8h;
typedef __attribute__((ext_vector_type(16))) _Float16 v16h;
typedef __attribute__((ext_vector_type(8)))  float    v8f;
typedef __attribute__((ext_vector_type(4)))  int      v4i;

// ---------------- CDNA5 async global->LDS copy ----------------
#if __has_builtin(__builtin_amdgcn_global_load_async_to_lds_b128)
#define HAS_ASYNC_LDS 1
#endif

__device__ __forceinline__ void cp16(h16* lds, const h16* g) {
#ifdef HAS_ASYNC_LDS
    typedef __attribute__((address_space(1))) v4i* gp_t;
    typedef __attribute__((address_space(3))) v4i* lp_t;
    __builtin_amdgcn_global_load_async_to_lds_b128(
        (gp_t)(uintptr_t)g, (lp_t)(unsigned)(uintptr_t)lds, 0, 0);
#else
    *(v8h*)lds = *(const v8h*)g;
#endif
}

__device__ __forceinline__ void wait_async() {
#if __has_builtin(__builtin_amdgcn_s_wait_asynccnt)
    __builtin_amdgcn_s_wait_asynccnt(0);
#else
    asm volatile("s_wait_asynccnt 0" ::: "memory");
#endif
}

__device__ __forceinline__ v8f wmma_f16(v16h a, v16h b, v8f c) {
    return __builtin_amdgcn_wmma_f32_16x16x32_f16(
        false, a, false, b, (short)0, c, false, false);
}

// halves [0..7] = p[0..7], halves [8..15] = p[16..23]; caller adds hi*8 (+ks*32)
__device__ __forceinline__ v16h load_frag(const h16* p) {
    union { v16h v; v8h h[2]; } u;
    u.h[0] = *(const v8h*)(p);
    u.h[1] = *(const v8h*)(p + 16);
    return u.v;
}

// ---------------- stage 0: converts ----------------
__global__ void cvt_f32_f16(const float* __restrict__ in, h16* __restrict__ out, int n4) {
    int i = blockIdx.x * blockDim.x + threadIdx.x;
    int stride = gridDim.x * blockDim.x;
    for (; i < n4; i += stride) {
        const float4 f = *(const float4*)(in + 4 * (size_t)i);
        v4h o; o[0] = (h16)f.x; o[1] = (h16)f.y; o[2] = (h16)f.z; o[3] = (h16)f.w;
        *(v4h*)(out + 4 * (size_t)i) = o;
    }
}

// in: [K][N] f32  ->  out: [N][K] f16 (transposed)
__global__ void cvt_transpose(const float* __restrict__ in, h16* __restrict__ out,
                              int K, int N) {
    int idx = blockIdx.x * blockDim.x + threadIdx.x;
    const int total = K * N;
    const int stride = gridDim.x * blockDim.x;
    for (; idx < total; idx += stride) {
        const int n = idx / K, k = idx - n * K;
        out[idx] = (h16)in[(size_t)k * N + n];
    }
}

// ---------------- shared GEMM mainloop ----------------
// A: [M][768] f16 row-major; BT: [Ncols][768] f16 (B transposed). acc = 2x4 wmma tiles.
__device__ __forceinline__ void stage_gemm(const h16* A, const h16* BT,
                                           h16 (*sA)[40], h16 (*sB)[40],
                                           int rowbase, int colbase, int kt, int tid) {
#pragma unroll
    for (int i = 0; i < 2; ++i) {
        const int flat = (i * 256 + tid) * 8;
        const int r = flat >> 5, k = flat & 31;
        cp16(&sA[r][k], A + (size_t)(rowbase + r) * DIMD + kt + k);
    }
#pragma unroll
    for (int i = 0; i < 2; ++i) {
        const int flat = (i * 256 + tid) * 8;
        const int n = flat >> 5, k = flat & 31;
        cp16(&sB[n][k], BT + (size_t)(colbase + n) * DIMD + kt + k);
    }
}

__device__ __forceinline__ void gemm_mainloop(const h16* A, const h16* BT,
                                              int rowbase, int colbase,
                                              v8f acc[2][4],
                                              h16 (*sA)[128][40], h16 (*sB)[128][40]) {
    const int tid = threadIdx.x;
    const int lane = tid & 31, lp = lane & 15, hi = lane >> 4;
    const int wid = tid >> 5, wm = wid & 3, wn = wid >> 2;

    stage_gemm(A, BT, sA[0], sB[0], rowbase, colbase, 0, tid);
    for (int t = 0; t < DIMD / 32; ++t) {
        wait_async();
        __syncthreads();
        if (t < DIMD / 32 - 1)
            stage_gemm(A, BT, sA[(t + 1) & 1], sB[(t + 1) & 1],
                       rowbase, colbase, (t + 1) * 32, tid);
        const int cb = t & 1;
        v16h aF[2], bF[4];
#pragma unroll
        for (int mt = 0; mt < 2; ++mt)
            aF[mt] = load_frag(&sA[cb][wm * 32 + mt * 16 + lp][hi * 8]);
#pragma unroll
        for (int nb = 0; nb < 4; ++nb)
            bF[nb] = load_frag(&sB[cb][wn * 64 + nb * 16 + lp][hi * 8]);
#pragma unroll
        for (int mt = 0; mt < 2; ++mt)
#pragma unroll
            for (int nb = 0; nb < 4; ++nb)
                acc[mt][nb] = wmma_f16(aF[mt], bF[nb], acc[mt][nb]);
    }
}

// ---------------- stage 1: QKV GEMM ----------------
__global__ __launch_bounds__(256) void qkv_gemm(const h16* __restrict__ A,
                                                const h16* __restrict__ BT,
                                                const float* __restrict__ bias,
                                                h16* __restrict__ qkvout) {
    __shared__ __align__(16) h16 sA[2][128][40];
    __shared__ __align__(16) h16 sB[2][128][40];
    const int tid = threadIdx.x;
    const int lane = tid & 31, lp = lane & 15, hi = lane >> 4;
    const int wid = tid >> 5, wm = wid & 3, wn = wid >> 2;
    const int rowbase = blockIdx.x * 128;
    const int colbase = blockIdx.y * 128;
    const int s = blockIdx.y / 6;                 // which of q/k/v (128 | 768)
    const int cbase = (blockIdx.y - s * 6) * 128; // column within the s-slab

    v8f acc[2][4] = {};
    gemm_mainloop(A, BT, rowbase, colbase, acc, sA, sB);

#pragma unroll
    for (int mt = 0; mt < 2; ++mt) {
#pragma unroll
        for (int nb = 0; nb < 4; ++nb) {
            const int colh = cbase + wn * 64 + nb * 16 + lp; // 0..767 within slab
            const int h = colh >> 6, d = colh & 63;
            const float bv = bias[s * DIMD + colh];
#pragma unroll
            for (int r = 0; r < 8; ++r) {
                const int grow = rowbase + wm * 32 + mt * 16 + r + 8 * hi;
                const int b = grow >> 10, n = grow & 1023;
                const int bh = b * NHEAD + h;
                const float v = acc[mt][nb][r] + bv;
                if (s == 2) {  // V stored transposed: [bh][hd][seq]
                    qkvout[2 * (size_t)HEADBUF + ((size_t)bh * HD + d) * SEQ + n] = (h16)v;
                } else {       // Q,K: [bh][seq][hd]
                    qkvout[(size_t)s * HEADBUF + ((size_t)bh * SEQ + n) * HD + d] = (h16)v;
                }
            }
        }
    }
}

// ---------------- stage 2: flash attention ----------------
// grid (16 q-chunks, 96 bh), 128 thr = 4 waves, each wave = 16 query rows
__device__ __forceinline__ void stage_kv(const h16* K, const h16* VT, size_t headoff,
                                         h16 (*sK)[72], h16 (*sV)[72], int kt, int tid) {
#pragma unroll
    for (int i = 0; i < 4; ++i) {
        const int flat = (i * 128 + tid) * 8;
        const int r = flat >> 6, c = flat & 63;
        cp16(&sK[r][c], K + headoff + (size_t)(kt + r) * HD + c);   // [kv][hd]
        cp16(&sV[r][c], VT + headoff + (size_t)r * SEQ + kt + c);   // [hd][kv]
    }
}

__global__ __launch_bounds__(128) void attn_kernel(const h16* __restrict__ qkv,
                                                   h16* __restrict__ wa) {
    __shared__ __align__(16) h16 sK[2][64][72];
    __shared__ __align__(16) h16 sV[2][64][72];
    __shared__ __align__(16) h16 sP[4][16][72];
    const int tid = threadIdx.x;
    const int w = tid >> 5, lane = tid & 31, lp = lane & 15, hi = lane >> 4;
    const int bh = blockIdx.y;
    const int b = bh / NHEAD, h = bh - b * NHEAD;
    const int q0 = blockIdx.x * 64 + w * 16;
    const size_t headoff = (size_t)bh * SEQ * HD;   // same for K and transposed V
    const h16* Q = qkv;
    const h16* K = qkv + HEADBUF;
    const h16* VT = qkv + 2 * (size_t)HEADBUF;

    v16h qf[2];
#pragma unroll
    for (int ks = 0; ks < 2; ++ks) {
        qf[ks] = load_frag(Q + headoff + (size_t)(q0 + lp) * HD + ks * 32 + hi * 8);
        qf[ks] = qf[ks] * (h16)SCALE;   // fold softmax scale into Q (exact: 2^-3)
    }

    v8f o[4] = {};
    float m[8], l[8];
#pragma unroll
    for (int r = 0; r < 8; ++r) { m[r] = -3.0e38f; l[r] = 0.0f; }

    stage_kv(K, VT, headoff, sK[0], sV[0], 0, tid);
    for (int t = 0; t < SEQ / 64; ++t) {
        wait_async();
        __syncthreads();
        if (t < SEQ / 64 - 1)
            stage_kv(K, VT, headoff, sK[(t + 1) & 1], sV[(t + 1) & 1], (t + 1) * 64, tid);
        const int cb = t & 1;

        // S = (Q*scale) * K^T for 16x64 tile
        v8f s4[4];
#pragma unroll
        for (int kb = 0; kb < 4; ++kb) {
            v8f s = {};
#pragma unroll
            for (int ks = 0; ks < 2; ++ks) {
                v16h kf = load_frag(&sK[cb][kb * 16 + lp][ks * 32 + hi * 8]);
                s = wmma_f16(qf[ks], kf, s);
            }
            s4[kb] = s;
        }

        // online softmax (row r of this lane = r + 8*hi; col = lp + kb*16)
#pragma unroll
        for (int r = 0; r < 8; ++r) {
            float rowm = fmaxf(fmaxf(s4[0][r], s4[1][r]), fmaxf(s4[2][r], s4[3][r]));
#pragma unroll
            for (int off = 1; off < 16; off <<= 1)
                rowm = fmaxf(rowm, __shfl_xor(rowm, off, 32));
            const float mn = fmaxf(m[r], rowm);
            const float alpha = __expf(m[r] - mn);
            m[r] = mn;
            float rs = 0.0f;
#pragma unroll
            for (int kb = 0; kb < 4; ++kb) {
                const float p = __expf(s4[kb][r] - mn);
                s4[kb][r] = p;
                rs += p;
            }
#pragma unroll
            for (int off = 1; off < 16; off <<= 1)
                rs += __shfl_xor(rs, off, 32);
            l[r] = l[r] * alpha + rs;
#pragma unroll
            for (int nb = 0; nb < 4; ++nb) o[nb][r] *= alpha;
        }

        // re-layout P through wave-private LDS into A fragments
#pragma unroll
        for (int kb = 0; kb < 4; ++kb)
#pragma unroll
            for (int r = 0; r < 8; ++r)
                sP[w][r + 8 * hi][kb * 16 + lp] = (h16)s4[kb][r];

        v16h pf[2];
#pragma unroll
        for (int ks = 0; ks < 2; ++ks)
            pf[ks] = load_frag(&sP[w][lp][ks * 32 + hi * 8]);
#pragma unroll
        for (int nb = 0; nb < 4; ++nb)
#pragma unroll
            for (int ks = 0; ks < 2; ++ks) {
                v16h vf = load_frag(&sV[cb][nb * 16 + lp][ks * 32 + hi * 8]);
                o[nb] = wmma_f16(pf[ks], vf, o[nb]);
            }
    }

    // finalize: O /= l, store to wa[B,N,D] (D index = h*64 + col)
    float linv[8];
#pragma unroll
    for (int r = 0; r < 8; ++r) linv[r] = 1.0f / l[r];
#pragma unroll
    for (int nb = 0; nb < 4; ++nb) {
        const int col = h * HD + nb * 16 + lp;
#pragma unroll
        for (int r = 0; r < 8; ++r) {
            const int grow = b * SEQ + q0 + r + 8 * hi;
            wa[(size_t)grow * DIMD + col] = (h16)(o[nb][r] * linv[r]);
        }
    }
}

// ---------------- stage 3: proj GEMM -> f32 out ----------------
__global__ __launch_bounds__(256) void proj_gemm(const h16* __restrict__ A,
                                                 const h16* __restrict__ BT,
                                                 const float* __restrict__ bias,
                                                 float* __restrict__ out) {
    __shared__ __align__(16) h16 sA[2][128][40];
    __shared__ __align__(16) h16 sB[2][128][40];
    const int tid = threadIdx.x;
    const int lane = tid & 31, lp = lane & 15, hi = lane >> 4;
    const int wid = tid >> 5, wm = wid & 3, wn = wid >> 2;
    const int rowbase = blockIdx.x * 128;
    const int colbase = blockIdx.y * 128;

    v8f acc[2][4] = {};
    gemm_mainloop(A, BT, rowbase, colbase, acc, sA, sB);

#pragma unroll
    for (int mt = 0; mt < 2; ++mt)
#pragma unroll
        for (int nb = 0; nb < 4; ++nb) {
            const int gc = colbase + wn * 64 + nb * 16 + lp;
            const float bv = bias[gc];
#pragma unroll
            for (int r = 0; r < 8; ++r) {
                const int grow = rowbase + wm * 32 + mt * 16 + r + 8 * hi;
                out[(size_t)grow * DIMD + gc] = acc[mt][nb][r] + bv;
            }
        }
}

// ---------------- launcher ----------------
extern "C" void kernel_launch(void* const* d_in, const int* in_sizes, int n_in,
                              void* d_out, int out_size, void* d_ws, size_t ws_size,
                              hipStream_t stream) {
    const float* x      = (const float*)d_in[0];
    const float* w_qkv  = (const float*)d_in[1];
    const float* b_qkv  = (const float*)d_in[2];
    const float* w_proj = (const float*)d_in[3];
    const float* b_proj = (const float*)d_in[4];
    float* out = (float*)d_out;

    h16* ws      = (h16*)d_ws;
    h16* x_h     = ws + OFF_XH;
    h16* wqkvT_h = ws + OFF_WQKV;    // [2304][768]
    h16* wprojT_h= ws + OFF_WPROJ;   // [768][768] transposed
    h16* qkv_h   = ws + OFF_QKV;
    h16* wa_h    = ws + OFF_WA;

    {   // stage 0: downcast / transpose inputs to f16
        int n4 = XH_ELEMS / 4;
        cvt_f32_f16<<<(n4 + 255) / 256, 256, 0, stream>>>(x, x_h, n4);
        cvt_transpose<<<(WQKV_ELEMS + 255) / 256, 256, 0, stream>>>(w_qkv, wqkvT_h, DIMD, NQKV);
        cvt_transpose<<<(WPROJ_ELEMS + 255) / 256, 256, 0, stream>>>(w_proj, wprojT_h, DIMD, DIMD);
    }

    // stage 1: QKV projection -> Q,K [bh][seq][hd], V [bh][hd][seq] (f16)
    qkv_gemm<<<dim3(MROWS / 128, NQKV / 128), 256, 0, stream>>>(x_h, wqkvT_h, b_qkv, qkv_h);

    // stage 2: attention -> wa (f16, [B,N,D])
    attn_kernel<<<dim3(SEQ / 64, BATCH * NHEAD), 128, 0, stream>>>(qkv_h, wa_h);

    // stage 3: output projection -> f32 out
    proj_gemm<<<dim3(MROWS / 128, DIMD / 128), 256, 0, stream>>>(wa_h, wprojT_h, b_proj, out);
}
